// FAGCN_91207925498520
// MI455X (gfx1250) — compile-verified, matrix operand order checked by reading, not run on hardware
//
#include <hip/hip_runtime.h>
#include <math.h>

// FAGCN: N=50000 nodes, E=800000 edges, F_IN=512, HID=128, K=4 layers, CLS=40
#define N_NODES 50000
#define N_EDGES 800000
#define F_IN    512
#define HID     128
#define N_LAYERS 4
#define N_CLS   40
#define EPS_C   0.3f

typedef __attribute__((ext_vector_type(16))) __bf16 v16bf;
typedef __attribute__((ext_vector_type(8)))  float  v8f;

union FragAB { v16bf v; unsigned u[8]; };
union FragC  { v8f  v; float e[8]; };

// f32 -> bf16 round-to-nearest-even (bit pattern)
__device__ __forceinline__ unsigned short f2bf(float f) {
  unsigned u = __builtin_bit_cast(unsigned, f);
  u += 0x7FFFu + ((u >> 16) & 1u);
  return (unsigned short)(u >> 16);
}

// ---------------------------------------------------------------------------
// deg / nd helpers
// ---------------------------------------------------------------------------
__global__ __launch_bounds__(256) void k_zero(float* __restrict__ p, int n) {
  int i = blockIdx.x * blockDim.x + threadIdx.x;
  if (i < n) p[i] = 0.f;
}

__global__ __launch_bounds__(256) void k_deg(const int* __restrict__ row,
                                             float* __restrict__ deg) {
  int e = blockIdx.x * blockDim.x + threadIdx.x;
  if (e < N_EDGES) atomicAdd(&deg[row[e]], 1.0f);
}

__global__ __launch_bounds__(256) void k_nd(float* __restrict__ nd) {
  int i = blockIdx.x * blockDim.x + threadIdx.x;
  if (i < N_NODES) {
    float d = nd[i];
    d = d < 1.0f ? 1.0f : d;
    nd[i] = rsqrtf(d);
  }
}

// ---------------------------------------------------------------------------
// f32 -> packed bf16x2 streaming conversion (one dword out per thread)
// ---------------------------------------------------------------------------
__global__ __launch_bounds__(256) void k_cvt_bf16(const float* __restrict__ src,
                                                  unsigned* __restrict__ dst,
                                                  int n2) {
  int i = blockIdx.x * blockDim.x + threadIdx.x;
  if (i < n2) {
    float a = src[2 * i], b = src[2 * i + 1];
    dst[i] = (unsigned)f2bf(a) | ((unsigned)f2bf(b) << 16);
  }
}

// ---------------------------------------------------------------------------
// GEMM1: h = relu(xb @ Wb^T + b1), xb:[N,512] bf16, Wb:[128,512] bf16.
// bf16 WMMA 16x16x32, f32 acc. Block: 128M x 128N, 8 waves (4M x 2N),
// wave tile 32M x 64N = 2x4 fragments. Tiles staged to LDS with
// global_load_async_to_lds_b128 (ASYNCcnt), double-buffered.
// LDS row pitch 80B: 16B-aligned for B128 async writes, bank-conflict-free
// for the 16-consecutive-row fragment reads (20 dwords, gcd(20,64)=4, rows
// 0..15 hit distinct banks).
// ---------------------------------------------------------------------------
#define LDS_PITCH_U32 20           // dwords per 32-bf16 row (16 data + 4 pad)
#define LDS_TILE_U32  (128 * LDS_PITCH_U32)

__global__ __launch_bounds__(256) void k_gemm1_relu(
    const unsigned short* __restrict__ xb, const unsigned short* __restrict__ wb,
    const float* __restrict__ b1, float* __restrict__ h) {
  __shared__ __align__(16) unsigned As32[2 * LDS_TILE_U32];
  __shared__ __align__(16) unsigned Bs32[2 * LDS_TILE_U32];

  const int t    = threadIdx.x;
  const int lane = t & 31;
  const int wave = t >> 5;
  const int wm   = wave & 3;    // wave position along M (4)
  const int wn   = wave >> 2;   // wave position along N (2)
  const int half = lane >> 4;   // ISA 16-bit A/B layout: lane half selects K block
  const int m16  = lane & 15;
  const int rowBase = blockIdx.x * 128;

  const unsigned ldsA = (unsigned)(unsigned long long)(void*)&As32[0];
  const unsigned ldsB = (unsigned)(unsigned long long)(void*)&Bs32[0];

  FragC acc[2][4];
#pragma unroll
  for (int m = 0; m < 2; ++m)
#pragma unroll
    for (int n = 0; n < 4; ++n)
#pragma unroll
      for (int q = 0; q < 8; ++q) acc[m][n].e[q] = 0.f;

  // Stage one 128x32 bf16 tile pair into LDS buffer `buf` via async B128 loads.
  // 512 x 16B chunks per tile; each of 256 threads issues 2 per tile.
  auto stage = [&](int buf, int k0) {
#pragma unroll
    for (int j = 0; j < 2; ++j) {
      int idx = t + j * 256;
      int r  = idx >> 2;        // tile row 0..127
      int c4 = idx & 3;         // 16B chunk within 64B row
      int gr = rowBase + r;
      if (gr >= N_NODES) gr = N_NODES - 1;  // clamp: padding rows never stored
      unsigned long long ga =
          (unsigned long long)(xb + (size_t)gr * F_IN + k0 + c4 * 8);
      unsigned la = ldsA + (unsigned)(buf * LDS_TILE_U32 * 4 + r * 80 + c4 * 16);
      asm volatile("global_load_async_to_lds_b128 %0, %1, off"
                   :: "v"(la), "v"(ga) : "memory");
      unsigned long long gw =
          (unsigned long long)(wb + (size_t)r * F_IN + k0 + c4 * 8);
      unsigned lb = ldsB + (unsigned)(buf * LDS_TILE_U32 * 4 + r * 80 + c4 * 16);
      asm volatile("global_load_async_to_lds_b128 %0, %1, off"
                   :: "v"(lb), "v"(gw) : "memory");
    }
  };

  stage(0, 0);
  asm volatile("s_wait_asynccnt 0x0" ::: "memory");
  __syncthreads();

  for (int i = 0; i < F_IN / 32; ++i) {
    const int p = i & 1;
    if (i + 1 < F_IN / 32) stage(1 - p, (i + 1) * 32);  // prefetch next tile

    const unsigned* Ab = &As32[p * LDS_TILE_U32];
    const unsigned* Bb = &Bs32[p * LDS_TILE_U32];

    // Fragment dword v holds K pair (kb, kb+1); kb = (v>>2)*16 + half*8 + 2*(v&3)
    // -> dword index kd = (v>>2)*8 + half*4 + (v&3). One LDS dword per frag reg.
    FragAB a[2], bfr[4];
#pragma unroll
    for (int m = 0; m < 2; ++m) {
      int row = wm * 32 + m * 16 + m16;
#pragma unroll
      for (int v = 0; v < 8; ++v) {
        int kd = (v >> 2) * 8 + half * 4 + (v & 3);
        a[m].u[v] = Ab[row * LDS_PITCH_U32 + kd];
      }
    }
#pragma unroll
    for (int n = 0; n < 4; ++n) {
      int col = wn * 64 + n * 16 + m16;  // B col n == W1 row n (W1 is [N,K])
#pragma unroll
      for (int v = 0; v < 8; ++v) {
        int kd = (v >> 2) * 8 + half * 4 + (v & 3);
        bfr[n].u[v] = Bb[col * LDS_PITCH_U32 + kd];
      }
    }

#pragma unroll
    for (int m = 0; m < 2; ++m)
#pragma unroll
      for (int n = 0; n < 4; ++n)
        acc[m][n].v = __builtin_amdgcn_wmma_f32_16x16x32_bf16(
            false, a[m].v, false, bfr[n].v, (short)0, acc[m][n].v, false, false);

    asm volatile("s_wait_asynccnt 0x0" ::: "memory");  // next tile landed
    __syncthreads();                                   // all waves done with p
  }

  // Epilogue: C/D layout -> M = q + 8*half, N = lane&15. Bias + ReLU.
#pragma unroll
  for (int m = 0; m < 2; ++m)
#pragma unroll
    for (int n = 0; n < 4; ++n) {
      int gc = wn * 64 + n * 16 + m16;
      float bias = b1[gc];
#pragma unroll
      for (int q = 0; q < 8; ++q) {
        int gr = rowBase + wm * 32 + m * 16 + q + 8 * half;
        if (gr < N_NODES) {
          float v = acc[m][n].e[q] + bias;
          h[(size_t)gr * HID + gc] = v > 0.f ? v : 0.f;
        }
      }
    }
}

// ---------------------------------------------------------------------------
// hnext = EPS * raw   (vectorized, N*HID divisible by 4)
// ---------------------------------------------------------------------------
__global__ __launch_bounds__(256) void k_init(const float4* __restrict__ raw,
                                              float4* __restrict__ hnext) {
  int i = blockIdx.x * blockDim.x + threadIdx.x;
  if (i < N_NODES * HID / 4) {
    float4 v = raw[i];
    hnext[i] = make_float4(EPS_C * v.x, EPS_C * v.y, EPS_C * v.z, EPS_C * v.w);
  }
}

// ---------------------------------------------------------------------------
// FA layer edge kernel: one wave32 per edge, float4 per lane (HID=128).
//   g    = tanh(h[r].gwa + h[c].gwb + gb)
//   norm = g * nd[r] * nd[c]
//   hout[c] += norm * h[r]   (atomic f32)
// ---------------------------------------------------------------------------
__global__ __launch_bounds__(256) void k_edge(
    const float* __restrict__ h, const int* __restrict__ row,
    const int* __restrict__ col, const float* __restrict__ nd,
    const float* __restrict__ gwa, const float* __restrict__ gwb,
    const float* __restrict__ gb, float* __restrict__ hout) {
  int gid  = blockIdx.x * blockDim.x + threadIdx.x;
  int e    = gid >> 5;
  int lane = threadIdx.x & 31;
  if (e >= N_EDGES) return;
  int r = row[e], c = col[e];
  const float4* hr4 = (const float4*)(h + (size_t)r * HID);
  const float4* hc4 = (const float4*)(h + (size_t)c * HID);
  const float4* wa4 = (const float4*)gwa;
  const float4* wb4 = (const float4*)gwb;
  float4 ar = hr4[lane];
  float4 ac = hc4[lane];
  float4 wa = wa4[lane];
  float4 wc = wb4[lane];
  float s = ar.x * wa.x + ar.y * wa.y + ar.z * wa.z + ar.w * wa.w
          + ac.x * wc.x + ac.y * wc.y + ac.z * wc.z + ac.w * wc.w;
#pragma unroll
  for (int off = 16; off > 0; off >>= 1) s += __shfl_xor(s, off, 32);
  float g    = tanhf(s + gb[0]);
  float norm = g * nd[r] * nd[c];
  float* dst = hout + (size_t)c * HID + lane * 4;
  atomicAdd(&dst[0], norm * ar.x);
  atomicAdd(&dst[1], norm * ar.y);
  atomicAdd(&dst[2], norm * ar.z);
  atomicAdd(&dst[3], norm * ar.w);
}

// ---------------------------------------------------------------------------
// out = log_softmax(h @ W2^T + b2)   one thread per node, 40 accumulators.
// W2 reads are wave-uniform -> scalar loads.
// ---------------------------------------------------------------------------
__global__ __launch_bounds__(256) void k_out(
    const float* __restrict__ h, const float* __restrict__ w2,
    const float* __restrict__ b2, float* __restrict__ out) {
  int node = blockIdx.x * blockDim.x + threadIdx.x;
  if (node >= N_NODES) return;
  float acc[N_CLS];
#pragma unroll
  for (int cc = 0; cc < N_CLS; ++cc) acc[cc] = b2[cc];
  const float* hn = h + (size_t)node * HID;
  for (int k = 0; k < HID; ++k) {
    float hk = hn[k];
#pragma unroll 8
    for (int cc = 0; cc < N_CLS; ++cc) acc[cc] += hk * w2[cc * HID + k];
  }
  float mx = acc[0];
#pragma unroll
  for (int cc = 1; cc < N_CLS; ++cc) mx = fmaxf(mx, acc[cc]);
  float sum = 0.f;
#pragma unroll
  for (int cc = 0; cc < N_CLS; ++cc) sum += expf(acc[cc] - mx);
  float lse = mx + logf(sum);
  float* o = out + (size_t)node * N_CLS;
#pragma unroll
  for (int cc = 0; cc < N_CLS; ++cc) o[cc] = acc[cc] - lse;
}

// ---------------------------------------------------------------------------
extern "C" void kernel_launch(void* const* d_in, const int* in_sizes, int n_in,
                              void* d_out, int out_size, void* d_ws, size_t ws_size,
                              hipStream_t stream) {
  const float* x   = (const float*)d_in[0];
  const int*   ei  = (const int*)  d_in[1];
  const float* t1w = (const float*)d_in[2];
  const float* t1b = (const float*)d_in[3];
  const float* gw  = (const float*)d_in[4];   // [4, 256]
  const float* gbv = (const float*)d_in[5];   // [4]
  const float* t2w = (const float*)d_in[6];
  const float* t2b = (const float*)d_in[7];
  float* out = (float*)d_out;

  const int* row = ei;
  const int* col = ei + N_EDGES;

  // Workspace (floats): nd[N] | raw | hA | hB | xb(bf16) | wb(bf16)  ~128.4 MB
  float* ws  = (float*)d_ws;
  float* nd  = ws;
  float* raw = ws + 50048;  // 256B-aligned past nd
  float* hA  = raw + (size_t)N_NODES * HID;
  float* hB  = hA  + (size_t)N_NODES * HID;
  unsigned short* xb = (unsigned short*)(hB + (size_t)N_NODES * HID);
  unsigned short* wb = xb + (size_t)N_NODES * F_IN;

  // degree -> nd = clip(deg,1)^-0.5
  k_zero<<<(N_NODES + 255) / 256, 256, 0, stream>>>(nd, N_NODES);
  k_deg <<<(N_EDGES + 255) / 256, 256, 0, stream>>>(row, nd);
  k_nd  <<<(N_NODES + 255) / 256, 256, 0, stream>>>(nd);

  // One streaming bf16 conversion pass for x and W1
  k_cvt_bf16<<<(N_NODES * F_IN / 2 + 255) / 256, 256, 0, stream>>>(
      x, (unsigned*)xb, N_NODES * F_IN / 2);
  k_cvt_bf16<<<(HID * F_IN / 2 + 255) / 256, 256, 0, stream>>>(
      t1w, (unsigned*)wb, HID * F_IN / 2);

  // h0 = relu(x @ W1^T + b1)  (bf16 WMMA + async-LDS double buffering)
  k_gemm1_relu<<<(N_NODES + 127) / 128, 256, 0, stream>>>(xb, wb, t1b, raw);

  // 4 FA layers, ping-pong hA/hB
  const float* hcur = raw;
  float* bufs[2] = {hA, hB};
  for (int k = 0; k < N_LAYERS; ++k) {
    float* hnext = bufs[k & 1];
    k_init<<<(N_NODES * HID / 4 + 255) / 256, 256, 0, stream>>>(
        (const float4*)raw, (float4*)hnext);
    k_edge<<<((N_EDGES * 32) + 255) / 256, 256, 0, stream>>>(
        hcur, row, col, nd, gw + k * 2 * HID, gw + k * 2 * HID + HID,
        gbv + k, hnext);
    hcur = hnext;
  }

  // log_softmax(h @ W2^T + b2)
  k_out<<<(N_NODES + 255) / 256, 256, 0, stream>>>(hcur, t2w, t2b, out);
}
